// Guide_17927193494452
// MI455X (gfx1250) — matrix operand order repaired and checked by reading, not compiled, not run on hardware
//
#include <hip/hip_runtime.h>
#include <stdint.h>

#define B_TOTAL 32768
#define OBS_DIM 100
#define EMB     16
#define HID     256
#define IN_DIM  132
#define MT      32      // batch rows per workgroup (2 row-tiles of 16)
#define KT_HH   8       // 256 / 32
#define KT_OBS  4       // 128 / 32 (obs padded 100->128)
#define HB_STRIDE  264  // bf16 elems; 528B rows -> rotates LDS banks
#define OBS_STRIDE 136
#define PAB_STRIDE 40

typedef __attribute__((ext_vector_type(16))) __bf16 v16bf;
typedef __attribute__((ext_vector_type(8)))  float  v8f;

union FragU { uint4 q[2]; v16bf v; };

// ---- workspace layout (bytes) ----
#define WS_WHH  0                        // 64*8*512 bf16 = 512KB (B-frag, gate-permuted)
#define WS_WOBS (512*1024)               // 64*4*512 bf16 = 256KB
#define WS_WPA  (WS_WOBS + 256*1024)     // 64*1*512 bf16 = 64KB  (W_ih[:,100:132])
#define WS_BIAS (WS_WPA + 64*1024)       // 1024 f32 (b_ih+b_hh, permuted)

__device__ __forceinline__ int permute_n(int np) {
    // wave w owns columns {i,f,g,o} x j in [32w, 32w+32)
    int w = np >> 7, within = np & 127;
    int nt = within >> 4, col = within & 15;
    int gate = nt >> 1;
    int j = w * 32 + (nt & 1) * 16 + col;
    return gate * 256 + j;   // original column in [i|f|g|o] concat order
}

__global__ void prep_weights(const float* __restrict__ W_ih, const float* __restrict__ W_hh,
                             const float* __restrict__ b_ih, const float* __restrict__ b_hh,
                             __bf16* __restrict__ whh, __bf16* __restrict__ wobs,
                             __bf16* __restrict__ wpa, float* __restrict__ bias) {
    int e = blockIdx.x * 256 + threadIdx.x;
    const int NWHH = 64 * 8 * 512, NWOBS = 64 * 4 * 512, NWPA = 64 * 512;
    if (e < NWHH) {
        int frag = e >> 9, within = e & 511;
        int lane = within >> 4, idx = within & 15;
        int kt = frag & 7, ntg = frag >> 3;
        int n = permute_n(ntg * 16 + (lane & 15));
        int k = kt * 32 + (lane >> 4) * 16 + idx;
        whh[e] = (__bf16)W_hh[n * HID + k];
        return;
    }
    e -= NWHH;
    if (e < NWOBS) {
        int frag = e >> 9, within = e & 511;
        int lane = within >> 4, idx = within & 15;
        int kt = frag & 3, ntg = frag >> 2;
        int n = permute_n(ntg * 16 + (lane & 15));
        int k = kt * 32 + (lane >> 4) * 16 + idx;
        wobs[e] = (__bf16)((k < OBS_DIM) ? W_ih[n * IN_DIM + k] : 0.f);
        return;
    }
    e -= NWOBS;
    if (e < NWPA) {
        int frag = e >> 9, within = e & 511;
        int lane = within >> 4, idx = within & 15;
        int n = permute_n(frag * 16 + (lane & 15));
        int k = (lane >> 4) * 16 + idx;                  // 0..31 -> x[100+k]
        wpa[e] = (__bf16)W_ih[n * IN_DIM + 100 + k];
        return;
    }
    e -= NWPA;
    if (e < 1024) {
        int n = permute_n(e);
        bias[e] = b_ih[n] + b_hh[n];
    }
}

__device__ __forceinline__ v8f wmma_bf16(v16bf a, v16bf b, v8f c) {
    return __builtin_amdgcn_wmma_f32_16x16x32_bf16(false, a, false, b, (short)0, c, false, false);
}

__device__ __forceinline__ v16bf load_a_lds(const __bf16* base, int stride, int kt, int lane) {
    int m = lane & 15, hi = lane >> 4;
    int k0 = kt * 32 + hi * 16;
    FragU f;
    f.q[0] = *(const uint4*)(base + m * stride + k0);
    f.q[1] = *(const uint4*)(base + m * stride + k0 + 8);
    return f.v;
}

__device__ __forceinline__ v16bf load_b_glb(const __bf16* __restrict__ base, int frag, int lane) {
    const __bf16* p = base + ((size_t)frag << 9) + (lane << 4);
    FragU f;
    f.q[0] = *(const uint4*)p;
    f.q[1] = *(const uint4*)(p + 8);
    return f.v;
}

__device__ __forceinline__ float sigm(float x)  { return 1.f / (1.f + __expf(-x)); }
__device__ __forceinline__ float ftanh(float x) { float e = __expf(2.f * x); return (e - 1.f) / (e + 1.f); }
__device__ __forceinline__ float gumbel(float u){ return -__logf(-__logf(u)); }

__global__ __launch_bounds__(256) void lstm_sampler(
    const float* __restrict__ obs,  const float* __restrict__ gprog,
    const float* __restrict__ gs0,  const float* __restrict__ gs1,
    const float* __restrict__ eps,
    const float* __restrict__ addr_emb, const float* __restrict__ prog_emb,
    const float* __restrict__ shape_emb,
    const float* __restrict__ Wprog, const float* __restrict__ bprog,
    const float* __restrict__ Wshape, const float* __restrict__ bshape,
    const float* __restrict__ Wpos,  const float* __restrict__ bpos,
    const float* __restrict__ Wpe,   const float* __restrict__ bpe,
    const __bf16* __restrict__ whh, const __bf16* __restrict__ wobs,
    const __bf16* __restrict__ wpa, const float* __restrict__ bias,
    float* __restrict__ out)
{
    __shared__ __align__(16) __bf16 hb[MT * HB_STRIDE];
    __shared__ __align__(16) __bf16 hsave[MT * HB_STRIDE];
    __shared__ __align__(16) __bf16 obsbf[MT * OBS_STRIDE];
    __shared__ __align__(16) __bf16 pab[MT * PAB_STRIDE];
    __shared__ float sc[MT * 8];
    __shared__ int   ids[MT * 4];     // [pid, shape_a, shape0, shape1]
    __shared__ float posst[MT * 8];   // [ax, ay, b0x, b0y, -,-,-,-]
    // cold per-lane state evicted from VGPRs (wave/lane private -> no barriers):
    __shared__ __align__(32) v8f zlds[8 * 2 * 8 * 32];  // z_obs seeds: 128KB
    __shared__ __align__(32) v8f clds[8 * 2 * 2 * 32];  // c1 snapshot: 32KB

    const int tid = threadIdx.x;
    const int wave = tid >> 5, lane = tid & 31;
    const int mlo = lane & 15, hi = lane >> 4;
    const int row0 = blockIdx.x * MT;

    // ---- stage obs (bf16, K padded to 128), zero h, pab = [0 | addr0] ----
    for (int i = tid; i < MT * OBS_STRIDE; i += 256) {
        int m = i / OBS_STRIDE, k = i - m * OBS_STRIDE;
        float v = (k < OBS_DIM) ? obs[(size_t)(row0 + m) * OBS_DIM + k] : 0.f;
        obsbf[i] = (__bf16)v;
    }
    for (int i = tid; i < MT * HB_STRIDE; i += 256) hb[i] = (__bf16)0.f;
    for (int i = tid; i < MT * PAB_STRIDE; i += 256) {
        int m = i / PAB_STRIDE, k = i - m * PAB_STRIDE;
        float v = (k >= EMB && k < 2 * EMB) ? addr_emb[k - EMB] : 0.f;  // addr id 0
        pab[i] = (__bf16)v;
    }
    __syncthreads();

    // ---- z_obs = obs @ Wobs^T + (b_ih+b_hh): computed once, parked in LDS ----
    // two row-tiles (rows 0-15, 16-31) share every B fragment -> 2x weight reuse
    v8f acc[2][8], cvec[2][2];
#pragma unroll
    for (int nt = 0; nt < 8; ++nt) {
        float bsc = bias[wave * 128 + nt * 16 + mlo];
#pragma unroll
        for (int r = 0; r < 8; ++r) { acc[0][nt][r] = bsc; acc[1][nt][r] = bsc; }
    }
#pragma unroll
    for (int kt = 0; kt < KT_OBS; ++kt) {
        v16bf a0 = load_a_lds(obsbf, OBS_STRIDE, kt, lane);
        v16bf a1 = load_a_lds(obsbf + 16 * OBS_STRIDE, OBS_STRIDE, kt, lane);
#pragma unroll
        for (int nt = 0; nt < 8; ++nt) {
            v16bf b = load_b_glb(wobs, (wave * 8 + nt) * 4 + kt, lane);
            acc[0][nt] = wmma_bf16(a0, b, acc[0][nt]);
            acc[1][nt] = wmma_bf16(a1, b, acc[1][nt]);
        }
    }
#pragma unroll
    for (int tl = 0; tl < 2; ++tl)
#pragma unroll
        for (int nt = 0; nt < 8; ++nt)
            zlds[((wave * 2 + tl) * 8 + nt) * 32 + lane] = acc[tl][nt];
#pragma unroll
    for (int tl = 0; tl < 2; ++tl)
#pragma unroll
        for (int t = 0; t < 2; ++t)
#pragma unroll
            for (int r = 0; r < 8; ++r) cvec[tl][t][r] = 0.f;

    // step ids: 0:h1(a0) 1:h2a(a1) 2:h3a(a2) 3:h2b(a3) 4:h3b(a4) 5:h4b(a5) 6:h5b(a6)
    for (int step = 0; step < 7; ++step) {
        // ---------- gate matmul: acc = z_obs + h@Whh' + [prev|addr]@Wpa' ----------
#pragma unroll
        for (int tl = 0; tl < 2; ++tl)
#pragma unroll
            for (int nt = 0; nt < 8; ++nt)
                acc[tl][nt] = zlds[((wave * 2 + tl) * 8 + nt) * 32 + lane];
#pragma unroll
        for (int kt = 0; kt < KT_HH; ++kt) {
            v16bf a0 = load_a_lds(hb, HB_STRIDE, kt, lane);
            v16bf a1 = load_a_lds(hb + 16 * HB_STRIDE, HB_STRIDE, kt, lane);
#pragma unroll
            for (int nt = 0; nt < 8; ++nt) {
                v16bf b = load_b_glb(whh, (wave * 8 + nt) * 8 + kt, lane);
                acc[0][nt] = wmma_bf16(a0, b, acc[0][nt]);
                acc[1][nt] = wmma_bf16(a1, b, acc[1][nt]);
            }
        }
        {
            v16bf a0 = load_a_lds(pab, PAB_STRIDE, 0, lane);
            v16bf a1 = load_a_lds(pab + 16 * PAB_STRIDE, PAB_STRIDE, 0, lane);
#pragma unroll
            for (int nt = 0; nt < 8; ++nt) {
                v16bf b = load_b_glb(wpa, wave * 8 + nt, lane);
                acc[0][nt] = wmma_bf16(a0, b, acc[0][nt]);
                acc[1][nt] = wmma_bf16(a1, b, acc[1][nt]);
            }
        }
        __syncthreads();   // all A-reads of hb/pab done before we overwrite hb

        // ---------- LSTM nonlinearity fully in registers (gate-permuted cols) ----
        // slice t: i=acc[t], f=acc[2+t], g=acc[4+t], o=acc[6+t]; j = 32w + 16t + mlo
#pragma unroll
        for (int tl = 0; tl < 2; ++tl) {
#pragma unroll
            for (int t = 0; t < 2; ++t) {
                v8f ci = cvec[tl][t];
                v8f cn, hn;
#pragma unroll
                for (int r = 0; r < 8; ++r) {
                    float iv = acc[tl][0 + t][r], fv = acc[tl][2 + t][r];
                    float gv = acc[tl][4 + t][r], ov = acc[tl][6 + t][r];
                    float c2 = sigm(fv) * ci[r] + sigm(iv) * ftanh(gv);
                    cn[r] = c2;
                    hn[r] = sigm(ov) * ftanh(c2);
                }
                cvec[tl][t] = cn;
                int jb = wave * 32 + t * 16 + mlo;
#pragma unroll
                for (int r = 0; r < 8; ++r)
                    hb[(tl * 16 + r + 8 * hi) * HB_STRIDE + jb] = (__bf16)hn[r];
            }
        }
        __syncthreads();   // h_new visible to heads + next step

        // ---------- sampling heads ----------
        if (step == 0) {
            if (tid < MT * 2) {               // prog logits
                int m = tid >> 1, p = tid & 1;
                float s = bprog[p]; const float* w = Wprog + p * HID;
                for (int k = 0; k < HID; ++k) s += (float)hb[m * HB_STRIDE + k] * w[k];
                sc[m * 8 + p] = s;
            }
            __syncthreads();
            if (tid < MT) {
                int row = row0 + tid;
                float v0 = sc[tid * 8 + 0] + gumbel(gprog[row * 2 + 0]);
                float v1 = sc[tid * 8 + 1] + gumbel(gprog[row * 2 + 1]);
                int bi = (v1 > v0) ? 1 : 0;
                ids[tid * 4 + 0] = bi;
                for (int j = 0; j < EMB; ++j) {
                    pab[tid * PAB_STRIDE + j]       = (__bf16)prog_emb[bi * EMB + j];
                    pab[tid * PAB_STRIDE + EMB + j] = (__bf16)addr_emb[1 * EMB + j];
                }
            }
            for (int i = tid; i < MT * HB_STRIDE; i += 256) hsave[i] = hb[i];
#pragma unroll
            for (int tl = 0; tl < 2; ++tl)
#pragma unroll
                for (int t = 0; t < 2; ++t)
                    clds[((wave * 2 + tl) * 2 + t) * 32 + lane] = cvec[tl][t];
        } else if (step == 1 || step == 3 || step == 4) {
            if (tid < MT * 3) {               // shape logits
                int m = tid / 3, p = tid - m * 3;
                float s = bshape[p]; const float* w = Wshape + p * HID;
                for (int k = 0; k < HID; ++k) s += (float)hb[m * HB_STRIDE + k] * w[k];
                sc[m * 8 + p] = s;
            }
            __syncthreads();
            const float* g = (step == 4) ? gs1 : gs0;
            int slot = (step == 1) ? 1 : (step == 3) ? 2 : 3;
            int na   = (step == 1) ? 2 : (step == 3) ? 4 : 5;
            if (tid < MT) {
                int row = row0 + tid;
                float best = -1e30f; int bi = 0;
                for (int p = 0; p < 3; ++p) {
                    float v = sc[tid * 8 + p] + gumbel(g[row * 3 + p]);
                    if (v > best) { best = v; bi = p; }
                }
                ids[tid * 4 + slot] = bi;
                for (int j = 0; j < EMB; ++j) {
                    pab[tid * PAB_STRIDE + j]       = (__bf16)shape_emb[bi * EMB + j];
                    pab[tid * PAB_STRIDE + EMB + j] = (__bf16)addr_emb[na * EMB + j];
                }
            }
        } else {                               // steps 2,5,6: position head
            if (tid < MT * 4) {
                int m = tid >> 2, p = tid & 3;
                float s = bpos[p]; const float* w = Wpos + p * HID;
                for (int k = 0; k < HID; ++k) s += (float)hb[m * HB_STRIDE + k] * w[k];
                sc[m * 8 + p] = s;
            }
            __syncthreads();
            if (step == 2) {
                if (tid < MT) {
                    int row = row0 + tid;
                    posst[tid * 8 + 0] = sc[tid * 8 + 0] + __expf(sc[tid * 8 + 2]) * eps[row * 4 + 0];
                    posst[tid * 8 + 1] = sc[tid * 8 + 1] + __expf(sc[tid * 8 + 3]) * eps[row * 4 + 1];
                }
                __syncthreads();               // head readers done; restore (h1,c1)
                for (int i = tid; i < MT * HB_STRIDE; i += 256) hb[i] = hsave[i];
#pragma unroll
                for (int tl = 0; tl < 2; ++tl)
#pragma unroll
                    for (int t = 0; t < 2; ++t)
                        cvec[tl][t] = clds[((wave * 2 + tl) * 2 + t) * 32 + lane];
                if (tid < MT) {
                    int bi = ids[tid * 4 + 0];
                    for (int j = 0; j < EMB; ++j) {
                        pab[tid * PAB_STRIDE + j]       = (__bf16)prog_emb[bi * EMB + j];
                        pab[tid * PAB_STRIDE + EMB + j] = (__bf16)addr_emb[3 * EMB + j];
                    }
                }
            } else if (step == 5) {
                if (tid < MT) {
                    int row = row0 + tid;
                    float x0 = sc[tid * 8 + 0] + __expf(sc[tid * 8 + 2]) * eps[row * 4 + 0];
                    float x1 = sc[tid * 8 + 1] + __expf(sc[tid * 8 + 3]) * eps[row * 4 + 1];
                    posst[tid * 8 + 2] = x0; posst[tid * 8 + 3] = x1;
                    for (int j = 0; j < EMB; ++j) {   // raw_position_embedder
                        float pe = bpe[j] + Wpe[j * 2 + 0] * x0 + Wpe[j * 2 + 1] * x1;
                        pab[tid * PAB_STRIDE + j]       = (__bf16)pe;
                        pab[tid * PAB_STRIDE + EMB + j] = (__bf16)addr_emb[6 * EMB + j];
                    }
                }
            } else { // step 6: finalize outputs
                if (tid < MT) {
                    int row = row0 + tid;
                    float y0 = sc[tid * 8 + 0] + __expf(sc[tid * 8 + 2]) * eps[row * 4 + 2];
                    float y1 = sc[tid * 8 + 1] + __expf(sc[tid * 8 + 3]) * eps[row * 4 + 3];
                    int pid = ids[tid * 4 + 0];
                    bool is0 = (pid == 0);
                    out[row] = (float)pid;
                    out[B_TOTAL + row * 2 + 0] = (float)(is0 ? ids[tid * 4 + 1] : ids[tid * 4 + 2]);
                    out[B_TOTAL + row * 2 + 1] = is0 ? 0.f : (float)ids[tid * 4 + 3];
                    float ax = posst[tid * 8 + 0], ay = posst[tid * 8 + 1];
                    float bx = posst[tid * 8 + 2], by = posst[tid * 8 + 3];
                    out[3 * B_TOTAL + row * 4 + 0] = is0 ? ax : bx;
                    out[3 * B_TOTAL + row * 4 + 1] = is0 ? ay : by;
                    out[3 * B_TOTAL + row * 4 + 2] = is0 ? 0.f : y0;
                    out[3 * B_TOTAL + row * 4 + 3] = is0 ? 0.f : y1;
                }
            }
        }
        __syncthreads();   // pab / hb stable before next iteration
    }
}

extern "C" void kernel_launch(void* const* d_in, const int* in_sizes, int n_in,
                              void* d_out, int out_size, void* d_ws, size_t ws_size,
                              hipStream_t stream) {
    const float* obs       = (const float*)d_in[0];
    const float* gprog     = (const float*)d_in[1];
    const float* gs0       = (const float*)d_in[2];
    const float* gs1       = (const float*)d_in[3];
    const float* eps       = (const float*)d_in[4];
    const float* W_ih      = (const float*)d_in[5];
    const float* W_hh      = (const float*)d_in[6];
    const float* b_ih      = (const float*)d_in[7];
    const float* b_hh      = (const float*)d_in[8];
    const float* addr_emb  = (const float*)d_in[9];
    const float* prog_emb  = (const float*)d_in[10];
    const float* shape_emb = (const float*)d_in[11];
    const float* Wprog     = (const float*)d_in[12];
    const float* bprog     = (const float*)d_in[13];
    const float* Wshape    = (const float*)d_in[14];
    const float* bshape    = (const float*)d_in[15];
    const float* Wpos      = (const float*)d_in[16];
    const float* bpos      = (const float*)d_in[17];
    const float* Wpe       = (const float*)d_in[18];
    const float* bpe       = (const float*)d_in[19];

    __bf16* whh  = (__bf16*)((char*)d_ws + WS_WHH);
    __bf16* wobs = (__bf16*)((char*)d_ws + WS_WOBS);
    __bf16* wpa  = (__bf16*)((char*)d_ws + WS_WPA);
    float*  bias = (float*)((char*)d_ws + WS_BIAS);

    // 262144 + 131072 + 32768 + 1024 = 427008 elements -> 1668 blocks of 256
    prep_weights<<<1668, 256, 0, stream>>>(W_ih, W_hh, b_ih, b_hh, whh, wobs, wpa, bias);

    lstm_sampler<<<B_TOTAL / MT, 256, 0, stream>>>(
        obs, gprog, gs0, gs1, eps,
        addr_emb, prog_emb, shape_emb,
        Wprog, bprog, Wshape, bshape, Wpos, bpos, Wpe, bpe,
        whh, wobs, wpa, bias, (float*)d_out);
}
